// ParagraphReader_68865505624605
// MI455X (gfx1250) — matrix operand order repaired and checked by reading, not compiled, not run on hardware
//
#include <hip/hip_runtime.h>
#include <hip/hip_bf16.h>
#include <math.h>

typedef _Float16 v8h  __attribute__((ext_vector_type(8)));
typedef _Float16 v16h __attribute__((ext_vector_type(16)));
typedef float    v8f  __attribute__((ext_vector_type(8)));

__device__ __forceinline__ float sigmf(float x) { return 1.f / (1.f + __expf(-x)); }

constexpr int BATCH = 16;
constexpr int H0 = 64, W0 = 256, CIN0 = 1,  C0 = 2;
constexpr int H1 = 32, W1 = 127, CIN1 = 6,  C1 = 10;
constexpr int H2 = 16, W2 = 62,  CIN2 = 20, C2 = 50;
constexpr int NOUT = 100;
constexpr int CPAD = 64;   // padded channel stride of block-2 f16 output

// ---------------------------------------------------------------------------
// 4-direction MD-LSTM wavefront. One workgroup per direction; diagonals are
// swept sequentially with workgroup barriers. Per cell:
//   a[16,5C] = [h_l | h_u | x] (16 x K)  @  [Wh; Wv; Wx] (K x 5C)   via WMMA
// Each wave owns a fixed 16-channel slice of all 5 gates; its B fragments are
// preloaded into registers once (weights are loop-invariant), so a diagonal
// step is just A-fragment LDS loads + a back-to-back WMMA chain + in-register
// sigmoid/tanh pointwise. State (h f16, c f32) lives in LDS, row-indexed.
// ---------------------------------------------------------------------------
template<int H, int W, int CIN, int C, int K, int CHUNK, bool F16OUT>
__global__ __launch_bounds__(512, 1)
void mdlstm_wavefront(const float* __restrict__ x,      // [B][CIN][H][W] (global coords)
                      const float* __restrict__ Wx,     // [4][CIN][5C]
                      const float* __restrict__ Wh,     // [4][C][5C]
                      const float* __restrict__ Wv,     // [4][C][5C]
                      const float* __restrict__ bias,   // [4][5C]
                      float* __restrict__ hout,         // [4][H][W][B][C]    (if !F16OUT)
                      _Float16* __restrict__ hout16)    // [4][H][W][B][CPAD] (if F16OUT)
{
    constexpr int NSL = (C + 15) / 16;   // 16-channel slices per gate
    constexpr int G5  = 5 * C;
    constexpr int NKS = K / 32;

    const int dir  = blockIdx.x;
    const int tid  = threadIdx.x, nthr = blockDim.x;
    const int lane = tid & 31, wid = tid >> 5, nw = nthr >> 5;

    __shared__ __align__(32) _Float16 Bt[5][NSL * 16][K];      // B^T tiles (gate, col, k)
    __shared__ __align__(32) _Float16 Arow[CHUNK][16][K];      // staged A rows per cell
    __shared__ _Float16 hst[2][H][BATCH][C];                   // h state, prev/cur diag
    __shared__ float    cst[2][H][BATCH][C];                   // c state
    __shared__ float    bs[G5];

    // --- load weights into LDS as f16, concatenated along K: [Wh; Wv; Wx]
    for (int idx = tid; idx < 5 * NSL * 16 * K; idx += nthr) {
        int k = idx % K, n = (idx / K) % (NSL * 16), g = idx / (K * NSL * 16);
        float v = 0.f;
        int col = g * C + n;
        if (n < C) {
            if (k < C)                 v = Wh[(dir * C + k) * G5 + col];
            else if (k < 2 * C)        v = Wv[(dir * C + (k - C)) * G5 + col];
            else if (k < 2 * C + CIN)  v = Wx[(dir * CIN + (k - 2 * C)) * G5 + col];
        }
        Bt[g][n][k] = (_Float16)v;
    }
    for (int idx = tid; idx < G5; idx += nthr) bs[idx] = bias[dir * G5 + idx];
    for (int idx = tid; idx < 2 * H * BATCH * C; idx += nthr) {
        (&hst[0][0][0][0])[idx] = (_Float16)0.f;
        (&cst[0][0][0][0])[idx] = 0.f;
    }
    __syncthreads();

    // --- fixed gate-slice per wave; preload its B fragments into registers
    const int sl    = wid % NSL;
    const int tstep = nw / NSL;
    v16h breg[5][NKS];
    #pragma unroll
    for (int g = 0; g < 5; ++g)
        #pragma unroll
        for (int ks = 0; ks < NKS; ++ks)
            breg[g][ks] = __builtin_shufflevector(
                *(const v8h*)&Bt[g][sl * 16 + (lane & 15)][ks * 32 + ((lane < 16) ? 0 : 8)],
                *(const v8h*)&Bt[g][sl * 16 + (lane & 15)][ks * 32 + 16 + ((lane < 16) ? 0 : 8)],
                0, 1, 2, 3, 4, 5, 6, 7, 8, 9, 10, 11, 12, 13, 14, 15);

    int prev = 0;
    for (int d = 0; d < H + W - 1; ++d) {
        const int ilo = (d > W - 1) ? d - (W - 1) : 0;
        const int ihi = (d < H - 1) ? d : H - 1;
        const int ncell = ihi - ilo + 1;

        for (int base = 0; base < ncell; base += CHUNK) {
            const int nc = (ncell - base < CHUNK) ? (ncell - base) : CHUNK;

            // --- stage packed A rows: [h_left | h_up | x_ij | 0pad]
            for (int idx = tid; idx < nc * 16 * K; idx += nthr) {
                int k = idx % K, m = (idx / K) & 15, t = idx / (16 * K);
                int i = ilo + base + t, j = d - i;
                float v = 0.f;
                if (k < C)            { if (j >= 1) v = (float)hst[prev][i][m][k]; }
                else if (k < 2 * C)   { if (i >= 1) v = (float)hst[prev][i - 1][m][k - C]; }
                else if (k < 2 * C + CIN) {
                    int gi = (dir & 2) ? (H - 1 - i) : i;
                    int gj = (dir & 1) ? (W - 1 - j) : j;
                    v = x[((m * CIN + (k - 2 * C)) * H + gi) * W + gj];
                }
                Arow[t][m][k] = (_Float16)v;
            }
            __syncthreads();

            // --- WMMA + pointwise; wave handles cells with stride tstep
            for (int t = wid / NSL; t < nc; t += tstep) {
                const int i = ilo + base + t, j = d - i;

                v8f acc[5] = {};
                #pragma unroll
                for (int ks = 0; ks < NKS; ++ks) {
                    const int off = (lane < 16) ? 0 : 8;
                    v8h alo = *(const v8h*)&Arow[t][lane & 15][ks * 32 + off];
                    v8h ahi = *(const v8h*)&Arow[t][lane & 15][ks * 32 + 16 + off];
                    v16h a = __builtin_shufflevector(alo, ahi,
                                0, 1, 2, 3, 4, 5, 6, 7, 8, 9, 10, 11, 12, 13, 14, 15);
                    #pragma unroll
                    for (int g = 0; g < 5; ++g)
                        acc[g] = __builtin_amdgcn_wmma_f32_16x16x32_f16(
                                     false, a, false, breg[g][ks], (short)0, acc[g],
                                     false, false);
                }

                const int n = lane & 15, ch = sl * 16 + n;
                const int gi = (dir & 2) ? (H - 1 - i) : i;
                const int gj = (dir & 1) ? (W - 1 - j) : j;
                if (ch < C) {
                    #pragma unroll
                    for (int r = 0; r < 8; ++r) {
                        const int m = r + ((lane >> 4) << 3);   // batch row
                        float ai  = acc[0][r] + bs[0 * C + ch];
                        float af1 = acc[1][r] + bs[1 * C + ch];
                        float af2 = acc[2][r] + bs[2 * C + ch];
                        float ao  = acc[3][r] + bs[3 * C + ch];
                        float ag  = acc[4][r] + bs[4 * C + ch];
                        float cl = (j >= 1) ? cst[prev][i][m][ch] : 0.f;
                        float cu = (i >= 1) ? cst[prev][i - 1][m][ch] : 0.f;
                        float c = sigmf(ai) * tanhf(ag) + sigmf(af1) * cl + sigmf(af2) * cu;
                        float h = sigmf(ao) * tanhf(c);
                        cst[prev ^ 1][i][m][ch] = c;
                        hst[prev ^ 1][i][m][ch] = (_Float16)h;
                        if constexpr (F16OUT)
                            hout16[(((size_t)(dir * H + gi) * W + gj) * BATCH + m) * CPAD + ch]
                                = (_Float16)h;
                        else
                            hout[(((size_t)(dir * H + gi) * W + gj) * BATCH + m) * C + ch] = h;
                    }
                }
            }
            __syncthreads();
        }
        prev ^= 1;
    }
}

// ---------------------------------------------------------------------------
// Sum 4 directions + strided 2x4 conv + tanh (tiny, VALU). Output in NCHW.
// ---------------------------------------------------------------------------
template<int H, int W, int CI, int O, int HO, int WO>
__global__ void conv_tanh(const float* __restrict__ hdir,  // [4][H][W][B][CI]
                          const float* __restrict__ w,     // [O][CI][2][4]
                          const float* __restrict__ cb,    // [O]
                          float* __restrict__ out)         // [B][O][HO][WO]
{
    int idx = blockIdx.x * blockDim.x + threadIdx.x;
    int total = BATCH * O * HO * WO;
    if (idx >= total) return;
    int j2 = idx % WO, i2 = (idx / WO) % HO, o = (idx / (WO * HO)) % O, b = idx / (WO * HO * O);
    float acc = cb[o];
    for (int ci = 0; ci < CI; ++ci)
        for (int kh = 0; kh < 2; ++kh)
            for (int kw = 0; kw < 4; ++kw) {
                int y = 2 * i2 + kh, xc = 2 * j2 + kw;
                float s = 0.f;
                #pragma unroll
                for (int d = 0; d < 4; ++d)
                    s += hdir[(((d * H + y) * W + xc) * BATCH + b) * CI + ci];
                acc += s * w[((o * CI + ci) * 2 + kh) * 4 + kw];
            }
    out[((b * O + o) * HO + i2) * WO + j2] = tanhf(acc);
}

// ---------------------------------------------------------------------------
__global__ void zero_half(_Float16* __restrict__ p, int n) {
    int i = blockIdx.x * blockDim.x + threadIdx.x;
    if (i < n) p[i] = (_Float16)0.f;
}

// ---------------------------------------------------------------------------
// Final: out[b,h,w,n] = sum_d h4[d,h,w,b,:] @ dW[d] + sum_d db[d].
// h4 is f16, channel-padded to 64: directions concatenate along K
// (4 x 64 = 256 -> 8 WMMA k-steps) with pure 16B vector loads; 7 N-tiles.
// ---------------------------------------------------------------------------
__global__ __launch_bounds__(256, 2)
void final_einsum(const _Float16* __restrict__ h16, // [4][H2][W2][B][CPAD]
                  const float* __restrict__ dW,     // [4][C2][NOUT]
                  const float* __restrict__ db,     // [4][NOUT]
                  float* __restrict__ out)          // [B][H2][W2][NOUT]
{
    __shared__ __align__(32) _Float16 dWt[7 * 16][256];
    __shared__ float dbs[7 * 16];
    const int tid = threadIdx.x, lane = tid & 31, wid = tid >> 5;

    for (int idx = tid; idx < 7 * 16 * 256; idx += blockDim.x) {
        int n = idx / 256, k = idx % 256;
        int d = k >> 6, c = k & 63;
        float v = (n < NOUT && c < C2) ? dW[(d * C2 + c) * NOUT + n] : 0.f;
        dWt[n][k] = (_Float16)v;
    }
    for (int idx = tid; idx < 7 * 16; idx += blockDim.x) {
        float s = 0.f;
        if (idx < NOUT)
            for (int d = 0; d < 4; ++d) s += db[d * NOUT + idx];
        dbs[idx] = s;
    }
    __syncthreads();

    const int cell = blockIdx.x * 8 + wid;
    if (cell < H2 * W2) {
        const int gi = cell / W2, gj = cell % W2;
        v8f acc[7] = {};
        #pragma unroll
        for (int ks = 0; ks < 8; ++ks) {
            const int d = ks >> 1, cb = (ks & 1) * 32;
            const int off = (lane < 16) ? 0 : 8;
            const _Float16* p = h16 +
                ((((size_t)(d * H2 + gi) * W2 + gj) * BATCH + (lane & 15)) * CPAD) + cb;
            v8h alo = *(const v8h*)&p[off];
            v8h ahi = *(const v8h*)&p[16 + off];
            v16h a = __builtin_shufflevector(alo, ahi,
                        0, 1, 2, 3, 4, 5, 6, 7, 8, 9, 10, 11, 12, 13, 14, 15);
            const int bk = ks * 32 + off;
            #pragma unroll
            for (int t = 0; t < 7; ++t) {
                v16h b = __builtin_shufflevector(
                    *(const v8h*)&dWt[t * 16 + (lane & 15)][bk],
                    *(const v8h*)&dWt[t * 16 + (lane & 15)][bk + 16],
                    0, 1, 2, 3, 4, 5, 6, 7, 8, 9, 10, 11, 12, 13, 14, 15);
                acc[t] = __builtin_amdgcn_wmma_f32_16x16x32_f16(
                             false, a, false, b, (short)0, acc[t], false, false);
            }
        }
        #pragma unroll
        for (int t = 0; t < 7; ++t) {
            const int n = t * 16 + (lane & 15);
            if (n < NOUT) {
                #pragma unroll
                for (int r = 0; r < 8; ++r) {
                    const int m = r + ((lane >> 4) << 3);
                    out[((m * H2 + gi) * W2 + gj) * NOUT + n] = acc[t][r] + dbs[n];
                }
            }
        }
    }
}

// ---------------------------------------------------------------------------
extern "C" void kernel_launch(void* const* d_in, const int* in_sizes, int n_in,
                              void* d_out, int out_size, void* d_ws, size_t ws_size,
                              hipStream_t stream) {
    (void)in_sizes; (void)n_in; (void)out_size; (void)ws_size;
    const float* x     = (const float*)d_in[0];
    const float* p0_Wx = (const float*)d_in[1];
    const float* p0_Wh = (const float*)d_in[2];
    const float* p0_Wv = (const float*)d_in[3];
    const float* p0_b  = (const float*)d_in[4];
    const float* c0_w  = (const float*)d_in[5];
    const float* c0_b  = (const float*)d_in[6];
    const float* p1_Wx = (const float*)d_in[7];
    const float* p1_Wh = (const float*)d_in[8];
    const float* p1_Wv = (const float*)d_in[9];
    const float* p1_b  = (const float*)d_in[10];
    const float* c1_w  = (const float*)d_in[11];
    const float* c1_b  = (const float*)d_in[12];
    const float* p2_Wx = (const float*)d_in[13];
    const float* p2_Wh = (const float*)d_in[14];
    const float* p2_Wv = (const float*)d_in[15];
    const float* p2_b  = (const float*)d_in[16];
    const float* dW    = (const float*)d_in[17];
    const float* db    = (const float*)d_in[18];

    char* ws = (char*)d_ws;
    auto alloc = [&](size_t bytes) { void* p = (void*)ws; ws += ((bytes + 255) / 256) * 256; return p; };
    float*    h0  = (float*)alloc((size_t)4 * H0 * W0 * BATCH * C0 * 4);       // 8.4 MB
    float*    x1  = (float*)alloc((size_t)BATCH * CIN1 * H1 * W1 * 4);         // 1.6 MB
    float*    h1  = (float*)alloc((size_t)4 * H1 * W1 * BATCH * C1 * 4);       // 10.4 MB
    float*    x2  = (float*)alloc((size_t)BATCH * CIN2 * H2 * W2 * 4);         // 1.3 MB
    _Float16* h2p = (_Float16*)alloc((size_t)4 * H2 * W2 * BATCH * CPAD * 2);  // 8.1 MB

    // block 0
    mdlstm_wavefront<H0, W0, CIN0, C0, 32, 16, false><<<4, 512, 0, stream>>>(
        x, p0_Wx, p0_Wh, p0_Wv, p0_b, h0, nullptr);
    {
        int total = BATCH * 6 * H1 * W1;
        conv_tanh<H0, W0, C0, 6, H1, W1><<<(total + 255) / 256, 256, 0, stream>>>(
            h0, c0_w, c0_b, x1);
    }
    // block 1
    mdlstm_wavefront<H1, W1, CIN1, C1, 32, 16, false><<<4, 512, 0, stream>>>(
        x1, p1_Wx, p1_Wh, p1_Wv, p1_b, h1, nullptr);
    {
        int total = BATCH * 20 * H2 * W2;
        conv_tanh<H1, W1, C1, 20, H2, W2><<<(total + 255) / 256, 256, 0, stream>>>(
            h1, c1_w, c1_b, x2);
    }
    // zero the padded f16 block-2 output (ws is poisoned; pad lanes must be 0)
    {
        int n = 4 * H2 * W2 * BATCH * CPAD;
        zero_half<<<(n + 255) / 256, 256, 0, stream>>>(h2p, n);
    }
    // block 2 (writes f16 channel-padded output)
    mdlstm_wavefront<H2, W2, CIN2, C2, 128, 8, true><<<4, 512, 0, stream>>>(
        x2, p2_Wx, p2_Wh, p2_Wv, p2_b, nullptr, h2p);
    // final per-direction linear + sum
    {
        int cells = H2 * W2;
        final_einsum<<<(cells + 7) / 8, 256, 0, stream>>>(h2p, dW, db, (float*)d_out);
    }
}
